// SpanNER_21028159881265
// MI455X (gfx1250) — matrix operand order repaired and checked by reading, not compiled, not run on hardware
//
#include <hip/hip_runtime.h>

typedef __attribute__((ext_vector_type(16))) _Float16 v16h;
typedef __attribute__((ext_vector_type(8)))  _Float16 v8h;
typedef __attribute__((ext_vector_type(4)))  _Float16 v4h;
typedef __attribute__((ext_vector_type(8)))  float    v8f;

#define SEQ    512
#define HID    1024
#define WD     64
#define DIN    2112
#define NSPAN  7575
#define NBATCH 8
#define NL     3
#define MT     128     // span rows per workgroup
#define NT     128     // H columns per pass
#define KT     32      // K chunk (f16 WMMA depth)
#define NKB    66      // DIN / KT
#define NNB    8       // HID / NT
#define TILES_PER_B 60 // ceil(7575/128)

// ---- dynamic LDS layout (bytes); As/Bs are double-buffered ----
#define As_STRIDE 40                                  // halves per A row (32 + pad)
#define Bs_STRIDE 40                                  // halves per B row (32 + pad)
#define Hs_STRIDE 133                                 // floats per h row (128 + pad)
#define As_HALVES (MT * As_STRIDE)                    // one A buffer, in halves
#define Bs_HALVES (NT * Bs_STRIDE)                    // one B buffer, in halves
#define As_OFF   0
#define Bs_OFF   (2 * As_HALVES * 2)                  // 20480
#define Hs_OFF   (Bs_OFF + 2 * Bs_HALVES * 2)         // 40960
#define W2_OFF   (Hs_OFF + MT * Hs_STRIDE * 4)        // 109056
#define B1_OFF   (W2_OFF + HID * NL * 4)              // 121344
#define LACC_OFF (B1_OFF + HID * 4)                   // 125440
#define SMEM_BYTES (LACC_OFF + MT * 2 * NL * 4)       // 128512

__global__ __launch_bounds__(256) void spanner_fused(
    const float* __restrict__ seq,   const float* __restrict__ wemb,
    const float* __restrict__ W1,    const float* __restrict__ b1,
    const float* __restrict__ W2,    const float* __restrict__ b2,
    const int*   __restrict__ starts,const int*   __restrict__ ends,
    float* __restrict__ out)
{
    extern __shared__ char smem[];
    _Float16* As  = (_Float16*)(smem + As_OFF);
    _Float16* Bs  = (_Float16*)(smem + Bs_OFF);
    float*    Hs  = (float*)(smem + Hs_OFF);
    float*    W2s = (float*)(smem + W2_OFF);
    float*    b1s = (float*)(smem + B1_OFF);
    float*    lac = (float*)(smem + LACC_OFF);

    const int tid  = threadIdx.x;
    const int lane = tid & 31;
    const int wid  = tid >> 5;
    const int wy   = wid >> 1;            // 0..3 : 32-row slice
    const int wx   = wid & 1;             // 0..1 : 64-col slice
    const int b    = blockIdx.x / TILES_PER_B;
    const int tile = blockIdx.x % TILES_PER_B;
    const int mBase = tile * MT;

    // preload W2 / b1 into LDS, zero logits accumulator
    for (int i = tid; i < HID * NL;    i += 256) W2s[i] = W2[i];
    for (int i = tid; i < HID;         i += 256) b1s[i] = b1[i];
    for (int i = tid; i < MT * 2 * NL; i += 256) lac[i] = 0.f;

    // per-thread gather geometry (4 rows: row = (tid>>3) + p*32)
    const int qa4    = (tid & 7) * 4;    // A: float offset in row
    const int n4     = (tid & 31) * 4;   // B: column quad
    const int kkBase = tid >> 5;         // B: kk = kkBase + p*8
    int    rowp[4];
    size_t seqS[4], seqE[4], wOff[4];
#pragma unroll
    for (int p = 0; p < 4; ++p) {
        int row = (tid >> 3) + p * 32;
        int mg  = mBase + row;
        int mi  = (mg < NSPAN) ? mg : 0;
        int s = starts[mi], e = ends[mi];
        rowp[p] = row;
        seqS[p] = ((size_t)(b * SEQ + s)) * HID;
        seqE[p] = ((size_t)(b * SEQ + e)) * HID;
        wOff[p] = (size_t)(e - s) * WD;
    }

    const int halfSel = (lane < 16) ? 0 : 1;
    const int moff    = (lane < 16) ? 0 : 8;
    const v8f vzero = {};

    __syncthreads();

    for (int nblk = 0; nblk < NNB; ++nblk) {
        const int nBase = nblk * NT;
        v8f acc[2][4];
#pragma unroll
        for (int i = 0; i < 2; ++i)
#pragma unroll
            for (int j = 0; j < 4; ++j) acc[i][j] = vzero;

        // ---------- staging helpers ----------
        auto load_chunk = [&](int kb, float4* va, float4* vb) {
#pragma unroll
            for (int p = 0; p < 4; ++p) {
                const float* src;
                if (kb < 32)      src = seq  + seqS[p] + kb * KT + qa4;
                else if (kb < 64) src = seq  + seqE[p] + (kb - 32) * KT + qa4;
                else              src = wemb + wOff[p] + (kb - 64) * KT + qa4;
                va[p] = *(const float4*)src;
            }
#pragma unroll
            for (int p = 0; p < 4; ++p) {
                int kk = kkBase + p * 8;
                vb[p] = *(const float4*)(W1 + (size_t)(kb * KT + kk) * HID + nBase + n4);
                if (kb + 1 < NKB)   // near-cache prefetch of the next W1 chunk
                    __builtin_prefetch(W1 + (size_t)((kb + 1) * KT + kk) * HID + nBase + n4, 0, 3);
            }
        };
        auto store_chunk = [&](int buf, const float4* va, const float4* vb) {
            _Float16* Ad = As + buf * As_HALVES;
            _Float16* Bd = Bs + buf * Bs_HALVES;
#pragma unroll
            for (int p = 0; p < 4; ++p) {
                v4h hv = { (_Float16)va[p].x, (_Float16)va[p].y,
                           (_Float16)va[p].z, (_Float16)va[p].w };
                *(v4h*)(Ad + rowp[p] * As_STRIDE + qa4) = hv;
            }
#pragma unroll
            for (int p = 0; p < 4; ++p) {
                int kk = kkBase + p * 8;
                Bd[(n4 + 0) * Bs_STRIDE + kk] = (_Float16)vb[p].x;
                Bd[(n4 + 1) * Bs_STRIDE + kk] = (_Float16)vb[p].y;
                Bd[(n4 + 2) * Bs_STRIDE + kk] = (_Float16)vb[p].z;
                Bd[(n4 + 3) * Bs_STRIDE + kk] = (_Float16)vb[p].w;
            }
        };

        // prologue: stage chunk 0 into buffer 0
        {
            float4 va[4], vb[4];
            load_chunk(0, va, vb);
            store_chunk(0, va, vb);
        }

        for (int kb = 0; kb < NKB; ++kb) {
            const int cur = kb & 1;
            __syncthreads();   // buf[cur] ready; everyone past compute of kb-1

            // issue next chunk's global loads BEFORE the WMMA block so the
            // loadcnt wait lands after ~12 ds_loads + 8 WMMAs of work
            float4 va[4], vb[4];
            const bool pre = (kb + 1 < NKB);
            if (pre) load_chunk(kb + 1, va, vb);

            // ---- fragments per ISA 16-bit layouts + WMMA ----
            const _Float16* Ab = As + cur * As_HALVES;
            const _Float16* Bb = Bs + cur * Bs_HALVES;
            v16h afrag[2];
#pragma unroll
            for (int i = 0; i < 2; ++i) {
                int m = wy * 32 + i * 16 + (lane & 15);
                const _Float16* ap = Ab + m * As_STRIDE + halfSel * 8;
                v8h lo = *(const v8h*)(ap);        // K = base .. base+7
                v8h hi = *(const v8h*)(ap + 16);   // K = 16+base .. 16+base+7
                afrag[i] = __builtin_shufflevector(lo, hi,
                    0,1,2,3,4,5,6,7,8,9,10,11,12,13,14,15);
            }
#pragma unroll
            for (int j = 0; j < 4; ++j) {
                int n = wx * 64 + j * 16 + (lane & 15);
                const _Float16* bp = Bb + n * Bs_STRIDE + halfSel * 8;
                v8h lo = *(const v8h*)(bp);        // K = base .. base+7
                v8h hi = *(const v8h*)(bp + 16);   // K = 16+base .. 16+base+7
                v16h bfrag = __builtin_shufflevector(lo, hi,
                    0,1,2,3,4,5,6,7,8,9,10,11,12,13,14,15);
#pragma unroll
                for (int i = 0; i < 2; ++i)
                    acc[i][j] = __builtin_amdgcn_wmma_f32_16x16x32_f16(
                        false, afrag[i], false, bfrag, (short)0, acc[i][j],
                        false, false);
            }

            // convert + store next chunk into the other buffer (made visible
            // to all waves by the barrier at the top of the next iteration)
            if (pre) store_chunk(1 - cur, va, vb);
        } // kb

        // ---- epilogue: bias + ReLU -> LDS h tile ----
#pragma unroll
        for (int i = 0; i < 2; ++i) {
#pragma unroll
            for (int j = 0; j < 4; ++j) {
                int nl = wx * 64 + j * 16 + (lane & 15);
                float bias = b1s[nBase + nl];
                int mrow = wy * 32 + i * 16 + moff;
#pragma unroll
                for (int r = 0; r < 8; ++r) {
                    float vv = acc[i][j][r] + bias;
                    Hs[(mrow + r) * Hs_STRIDE + nl] = vv > 0.f ? vv : 0.f;
                }
            }
        }
        __syncthreads();

        // ---- tiny GEMM2: h[128x128] @ W2[128x3], accumulated in LDS ----
        {
            int ml = tid & 127, hf = tid >> 7;
            const float* hrow = Hs + ml * Hs_STRIDE + hf * 64;
            const float* w2p  = W2s + (nBase + hf * 64) * NL;
            float s0 = 0.f, s1 = 0.f, s2 = 0.f;
#pragma unroll 8
            for (int n = 0; n < 64; ++n) {
                float hv = hrow[n];
                s0 += hv * w2p[n * 3 + 0];
                s1 += hv * w2p[n * 3 + 1];
                s2 += hv * w2p[n * 3 + 2];
            }
            float* lp = lac + (ml * 2 + hf) * NL;
            lp[0] += s0; lp[1] += s1; lp[2] += s2;
        }
        __syncthreads();
    } // nblk

    // ---- final: combine halves, add b2, store logits ----
    for (int idx = tid; idx < MT * NL; idx += 256) {
        int ml = idx / NL, l = idx % NL;
        int mg = mBase + ml;
        if (mg < NSPAN) {
            float v = lac[(ml * 2) * NL + l] + lac[(ml * 2 + 1) * NL + l] + b2[l];
            out[((size_t)b * NSPAN + mg) * NL + l] = v;
        }
    }
}

extern "C" void kernel_launch(void* const* d_in, const int* in_sizes, int n_in,
                              void* d_out, int out_size, void* d_ws, size_t ws_size,
                              hipStream_t stream) {
    const float* seq    = (const float*)d_in[0];
    const float* wemb   = (const float*)d_in[1];
    const float* W1     = (const float*)d_in[2];
    const float* b1     = (const float*)d_in[3];
    const float* W2     = (const float*)d_in[4];
    const float* b2     = (const float*)d_in[5];
    const int*   starts = (const int*)d_in[6];
    const int*   ends   = (const int*)d_in[7];
    float*       out    = (float*)d_out;

    (void)in_sizes; (void)n_in; (void)out_size; (void)d_ws; (void)ws_size;

    // allow >64KB dynamic LDS (host-side attribute, safe under graph capture)
    (void)hipFuncSetAttribute((const void*)spanner_fused,
                              hipFuncAttributeMaxDynamicSharedMemorySize,
                              SMEM_BYTES);

    spanner_fused<<<NBATCH * TILES_PER_B, 256, SMEM_BYTES, stream>>>(
        seq, wemb, W1, b1, W2, b2, starts, ends, out);
}
